// Transformer_77515569758848
// MI455X (gfx1250) — compile-verified, matrix operand order checked by reading, not run on hardware
//
#include <hip/hip_runtime.h>
#include <hip/hip_bf16.h>

typedef __bf16 bf16;
typedef __attribute__((ext_vector_type(4)))  bf16  v4bf;
typedef __attribute__((ext_vector_type(8)))  bf16  v8bf;
typedef __attribute__((ext_vector_type(16))) bf16  v16bf;
typedef __attribute__((ext_vector_type(8)))  float v8f;

constexpr int kVocab = 50257;
constexpr int kSeq   = 2048;
constexpr int kEmb   = 512;
constexpr int kAttn  = 64;
constexpr int kBatch = 8;
constexpr int kCtx   = 4;
constexpr int kRowsQ = kBatch * kSeq;              // 16384
constexpr int kRowsC = kCtx * kBatch * kSeq;       // 65536
constexpr int kRowsAll = kRowsQ + kRowsC;          // 81920

// ---------------------------------------------------------------- helpers

__device__ __forceinline__ v16bf cat16(v8bf lo, v8bf hi) {
  return __builtin_shufflevector(lo, hi, 0,1,2,3,4,5,6,7,8,9,10,11,12,13,14,15);
}

__device__ __forceinline__ v8f wmma_bf16(v16bf a, v16bf b, v8f c) {
  // D = A(16x32 bf16) * B(32x16 bf16) + C(16x16 f32)
  return __builtin_amdgcn_wmma_f32_16x16x32_bf16(false, a, false, b, (short)0, c,
                                                 false, false);
}

// A/B fragment loader for a row-major 16-bit operand: lane holds one row
// (A) or one column's source row (B); K chunks {kb+8*half .. +8} and {+16}.
__device__ __forceinline__ v16bf load_frag(const bf16* rowp, int kb, int half) {
  const int c0 = kb + half * 8;
  v8bf lo = *(const v8bf*)(rowp + c0);
  v8bf hi = *(const v8bf*)(rowp + c0 + 16);
  return cat16(lo, hi);
}

__device__ __forceinline__ float halfmax(float v) {
  v = fmaxf(v, __shfl_xor(v, 1, 32));
  v = fmaxf(v, __shfl_xor(v, 2, 32));
  v = fmaxf(v, __shfl_xor(v, 4, 32));
  v = fmaxf(v, __shfl_xor(v, 8, 32));
  return v;
}
__device__ __forceinline__ float halfsum(float v) {
  v += __shfl_xor(v, 1, 32);
  v += __shfl_xor(v, 2, 32);
  v += __shfl_xor(v, 4, 32);
  v += __shfl_xor(v, 8, 32);
  return v;
}

// ---------------------------------------------------------------- kernel 1:
// W_emb [E,V] f32 -> W_embT [V,E] bf16 (LDS-tiled transpose, coalesced both sides)
__global__ __launch_bounds__(256) void transpose_cast_kernel(
    const float* __restrict__ W, bf16* __restrict__ WT) {
  __shared__ float tile[32][33];
  const int v0 = blockIdx.x * 32;
  const int e0 = blockIdx.y * 32;
#pragma unroll
  for (int i = 0; i < 4; ++i) {
    int e = e0 + threadIdx.y + i * 8;
    int v = v0 + threadIdx.x;
    float val = 0.f;
    if (v < kVocab) val = W[(size_t)e * kVocab + v];
    tile[threadIdx.y + i * 8][threadIdx.x] = val;
  }
  __syncthreads();
#pragma unroll
  for (int i = 0; i < 4; ++i) {
    int vl = threadIdx.y + i * 8;
    int v  = v0 + vl;
    int e  = e0 + threadIdx.x;
    if (v < kVocab) WT[(size_t)v * kEmb + e] = (bf16)tile[threadIdx.x][vl];
  }
}

// ---------------------------------------------------------------- kernel 2:
// f32 -> bf16 elementwise cast (projection weights)
__global__ __launch_bounds__(256) void cast_kernel(const float* __restrict__ in,
                                                   bf16* __restrict__ out, int n) {
  int i = blockIdx.x * blockDim.x + threadIdx.x;
  if (i < n) out[i] = (bf16)in[i];
}

// ---------------------------------------------------------------- kernel 3:
// Materialize bf16 embeddings once:
//   emb[row] = bf16( W_embT[tok[row]] + W_pos[row % S] )
// rows [0, 16384)  <- tokens,  rows [16384, 81920) <- context.
// Coalesced 8B loads/stores; the bf16 conversion VALU happens exactly once.
__global__ __launch_bounds__(256) void embed_kernel(
    const int* __restrict__ toks, const int* __restrict__ ctx,
    const bf16* __restrict__ WT, const float* __restrict__ pos,
    bf16* __restrict__ emb) {
  const int t   = threadIdx.x & 127;              // 128 threads per row
  const int row = blockIdx.x * 2 + (threadIdx.x >> 7);
  const int tok = (row < kRowsQ) ? toks[row] : ctx[row - kRowsQ];
  const int s   = row & (kSeq - 1);
  const int e   = t * 4;
  v4bf tv = *(const v4bf*)(WT + (size_t)tok * kEmb + e);
  float4 p = *(const float4*)(pos + (size_t)s * kEmb + e);
  v4bf r;
  r[0] = (bf16)((float)tv[0] + p.x);
  r[1] = (bf16)((float)tv[1] + p.y);
  r[2] = (bf16)((float)tv[2] + p.z);
  r[3] = (bf16)((float)tv[3] + p.w);
  *(v4bf*)(emb + (size_t)row * kEmb + e) = r;
}

// ---------------------------------------------------------------- kernel 4:
// Register-blocked projection GEMM: each wave computes a 16x64 output slab
// (4 N-tiles, 32 accumulator VGPRs); the A fragment is loaded once per K-step
// and reused by 4 WMMAs.
//   Y = bf16( X @ W^T + bias ),  X bf16 [M,512] row-major, W bf16 [N,512].
// trans==0: Y row-major [M,N].  trans==1: Y transposed [N,M] (for V).
__global__ __launch_bounds__(256) void gemm_kernel(
    const bf16* __restrict__ X, const bf16* __restrict__ Wn,
    const float* __restrict__ bias, bf16* __restrict__ y,
    int M, int N, int trans) {
  const int lane = threadIdx.x & 31;
  const int w    = threadIdx.x >> 5;
  const int half = lane >> 4;
  const int mr   = lane & 15;

  const int ngroups = N >> 6;                 // 64-wide N groups
  const int gw = blockIdx.x * 8 + w;
  const int mt = gw / ngroups;
  const int ng = gw % ngroups;
  if (mt * 16 >= M) return;

  const bf16* arow = X + (size_t)(mt * 16 + mr) * kEmb;
  const bf16* brow[4];
  float bv[4];
#pragma unroll
  for (int nt = 0; nt < 4; ++nt) {
    const int n = ng * 64 + nt * 16 + mr;
    brow[nt] = Wn + (size_t)n * kEmb;         // column n of W^T == row n of W
    bv[nt]   = bias[n];
  }

  v8f zero = {};
  v8f acc[4];
#pragma unroll
  for (int nt = 0; nt < 4; ++nt) acc[nt] = zero;

#pragma unroll 2
  for (int ks = 0; ks < kEmb / 32; ++ks) {
    const int c0 = ks * 32 + half * 8;
    // issue all loads for this K-step before the WMMA quartet
    v8bf a0 = *(const v8bf*)(arow + c0);
    v8bf a1 = *(const v8bf*)(arow + c0 + 16);
    v8bf b0[4], b1[4];
#pragma unroll
    for (int nt = 0; nt < 4; ++nt) {
      b0[nt] = *(const v8bf*)(brow[nt] + c0);
      b1[nt] = *(const v8bf*)(brow[nt] + c0 + 16);
    }
    if (ks + 1 < kEmb / 32) __builtin_prefetch(arow + c0 + 32, 0, 1);
    v16bf A = cat16(a0, a1);
#pragma unroll
    for (int nt = 0; nt < 4; ++nt)
      acc[nt] = wmma_bf16(A, cat16(b0[nt], b1[nt]), acc[nt]);
  }

#pragma unroll
  for (int nt = 0; nt < 4; ++nt) {
#pragma unroll
    for (int i = 0; i < 8; ++i) acc[nt][i] += bv[nt];
  }

  if (!trans) {
#pragma unroll
    for (int nt = 0; nt < 4; ++nt) {
      const int n = ng * 64 + nt * 16 + mr;
#pragma unroll
      for (int i = 0; i < 8; ++i)
        y[(size_t)(mt * 16 + i + 8 * half) * N + n] = (bf16)acc[nt][i];
    }
  } else {
#pragma unroll
    for (int nt = 0; nt < 4; ++nt) {
      const int n = ng * 64 + nt * 16 + mr;
      v8bf o;
#pragma unroll
      for (int i = 0; i < 8; ++i) o[i] = (bf16)acc[nt][i];
      // rows m = mt*16 + 8*half + i are consecutive -> one 16B store per lane
      *(v8bf*)(y + (size_t)n * M + mt * 16 + 8 * half) = o;
    }
  }
}

// ---------------------------------------------------------------- kernel 5:
// Two-pass flash attention. One workgroup (8 waves) per (c,b, 16-query tile).
// Pass 1: per-wave online (rowmax, rowsum) over its 256 keys, merged in LDS.
// Pass 2: recompute scores, P=exp(s-m)/l -> LDS (bf16), then P @ V via WMMA
// with V read from the transposed buffer vt[E][C*B*S] (contiguous b128 loads,
// next chunk prefetched).
__global__ __launch_bounds__(256) void attn_kernel(
    const bf16* __restrict__ qb,   // [B*S, 64]
    const bf16* __restrict__ kb,   // [C*B*S, 64]
    const bf16* __restrict__ vt,   // [E=512, C*B*S]
    float* __restrict__ out) {     // [C,B,S,E]
  __shared__ __align__(16) bf16 p_sh[16][136];
  __shared__ float wmaxs[8][16];
  __shared__ float wsums[8][16];
  __shared__ float rowm[16];
  __shared__ float rowl[16];

  const int tid  = threadIdx.x;
  const int lane = tid & 31;
  const int w    = tid >> 5;
  const int half = lane >> 4;
  const int mr   = lane & 15;

  const int gid = blockIdx.x;
  const int qt  = gid & 127;       // S/16 = 128 query tiles
  const int cb  = gid >> 7;        // c*B + b
  const int b   = cb & (kBatch - 1);
  const int q0  = qt * 16;

  const size_t kbase = (size_t)cb * kSeq;
  const float  scale = 0.125f;     // 1/sqrt(64)

  // Q fragments (A operand, reused for every key chunk): lane row = q0+mr
  const bf16* qrow = qb + ((size_t)b * kSeq + q0 + mr) * kAttn;
  const v16bf QA0 = load_frag(qrow, 0, half);
  const v16bf QA1 = load_frag(qrow, 32, half);

  // ---- pass 1: per-row running max / sum-exp (this wave covers keys
  //      t = j0 + w*16 .. +16 for j0 stepping by 128)
  float mrun[8], lrun[8];
#pragma unroll
  for (int r = 0; r < 8; ++r) { mrun[r] = -__builtin_inff(); lrun[r] = 0.f; }

  for (int j0 = 0; j0 < kSeq; j0 += 128) {
    const bf16* krow = kb + (kbase + j0 + w * 16 + mr) * kAttn;
    if (j0 + 128 < kSeq) __builtin_prefetch(krow + 128 * kAttn, 0, 1);
    v8f sc = {};
    sc = wmma_bf16(QA0, load_frag(krow, 0, half), sc);
    sc = wmma_bf16(QA1, load_frag(krow, 32, half), sc);
#pragma unroll
    for (int r = 0; r < 8; ++r) {
      float x    = sc[r] * scale;
      float cmax = halfmax(x);
      float mnew = fmaxf(mrun[r], cmax);
      float csum = halfsum(__expf(x - mnew));
      lrun[r] = lrun[r] * __expf(mrun[r] - mnew) + csum;
      mrun[r] = mnew;
    }
  }
  if (mr == 0) {
#pragma unroll
    for (int r = 0; r < 8; ++r) {
      wmaxs[w][r + 8 * half] = mrun[r];
      wsums[w][r + 8 * half] = lrun[r];
    }
  }
  __syncthreads();
  if (tid < 16) {
    float m = -__builtin_inff();
#pragma unroll
    for (int ww = 0; ww < 8; ++ww) m = fmaxf(m, wmaxs[ww][tid]);
    float l = 0.f;
#pragma unroll
    for (int ww = 0; ww < 8; ++ww) l += wsums[ww][tid] * __expf(wmaxs[ww][tid] - m);
    rowm[tid] = m;
    rowl[tid] = l;
  }
  __syncthreads();

  float rm[8], rli[8];
#pragma unroll
  for (int r = 0; r < 8; ++r) {
    rm[r]  = rowm[r + 8 * half];
    rli[r] = 1.0f / rowl[r + 8 * half];
  }

  // ---- pass 2: recompute scores, normalized P -> LDS, then P @ V
  v8f zero = {};
  v8f accv[4];
#pragma unroll
  for (int nt = 0; nt < 4; ++nt) accv[nt] = zero;

  for (int j0 = 0; j0 < kSeq; j0 += 128) {
    const bf16* krow = kb + (kbase + j0 + w * 16 + mr) * kAttn;
    v8f sc = {};
    sc = wmma_bf16(QA0, load_frag(krow, 0, half), sc);
    sc = wmma_bf16(QA1, load_frag(krow, 32, half), sc);
#pragma unroll
    for (int r = 0; r < 8; ++r) {
      float p = __expf(sc[r] * scale - rm[r]) * rli[r];
      p_sh[r + 8 * half][w * 16 + mr] = (bf16)p;  // D row r+8h, col w*16+mr
    }
    __syncthreads();

    const int e0 = w * 64;                        // this wave's 64 output cols
#pragma unroll
    for (int nt = 0; nt < 4; ++nt) {
      const int e = e0 + nt * 16 + mr;            // B column for this lane
      const bf16* vcol = vt + (size_t)e * kRowsC + kbase + j0;
      if (j0 + 128 < kSeq) __builtin_prefetch(vcol + 128, 0, 1);
#pragma unroll
      for (int kk = 0; kk < 4; ++kk) {
        const int c0 = kk * 32 + half * 8;
        v8bf a0 = *(const v8bf*)&p_sh[mr][c0];
        v8bf a1 = *(const v8bf*)&p_sh[mr][c0 + 16];
        v8bf b0 = *(const v8bf*)(vcol + c0);
        v8bf b1 = *(const v8bf*)(vcol + c0 + 16);
        accv[nt] = wmma_bf16(cat16(a0, a1), cat16(b0, b1), accv[nt]);
      }
    }
    __syncthreads();
  }

  // store f32 output [C,B,S,E]
  const size_t obase = (kbase + q0) * (size_t)kEmb;
#pragma unroll
  for (int nt = 0; nt < 4; ++nt) {
    const int e = w * 64 + nt * 16 + mr;
#pragma unroll
    for (int i = 0; i < 8; ++i)
      out[obase + (size_t)(i + 8 * half) * kEmb + e] = accv[nt][i];
  }
}

// ---------------------------------------------------------------- launcher

extern "C" void kernel_launch(void* const* d_in, const int* in_sizes, int n_in,
                              void* d_out, int out_size, void* d_ws, size_t ws_size,
                              hipStream_t stream) {
  const int*   tokens  = (const int*)d_in[0];
  const int*   context = (const int*)d_in[1];
  const float* W_emb = (const float*)d_in[2];
  const float* W_pos = (const float*)d_in[3];
  const float* W_q   = (const float*)d_in[4];
  const float* b_q   = (const float*)d_in[5];
  const float* W_k   = (const float*)d_in[6];
  const float* b_k   = (const float*)d_in[7];
  const float* W_v   = (const float*)d_in[8];
  const float* b_v   = (const float*)d_in[9];
  float* out = (float*)d_out;
  (void)in_sizes; (void)n_in; (void)out_size; (void)ws_size;

  char* ws = (char*)d_ws;
  size_t off = 0;
  auto alloc = [&](size_t bytes) {
    size_t o = off;
    off = (off + bytes + 255) & ~(size_t)255;
    return o;
  };
  bf16* wEmbT = (bf16*)(ws + alloc((size_t)kVocab * kEmb * 2));
  bf16* wqB   = (bf16*)(ws + alloc((size_t)kAttn * kEmb * 2));
  bf16* wkB   = (bf16*)(ws + alloc((size_t)kAttn * kEmb * 2));
  bf16* wvB   = (bf16*)(ws + alloc((size_t)kEmb * kEmb * 2));
  bf16* embA  = (bf16*)(ws + alloc((size_t)kRowsAll * kEmb * 2));
  bf16* qB    = (bf16*)(ws + alloc((size_t)kRowsQ * kAttn * 2));
  bf16* kB    = (bf16*)(ws + alloc((size_t)kRowsC * kAttn * 2));
  bf16* vT    = (bf16*)(ws + alloc((size_t)kEmb * kRowsC * 2));
  const bf16* embQ = embA;
  const bf16* embC = embA + (size_t)kRowsQ * kEmb;

  // 1. transpose + cast embedding table (one-time; makes all gathers coalesced)
  transpose_cast_kernel<<<dim3((kVocab + 31) / 32, kEmb / 32), dim3(32, 8), 0,
                          stream>>>(W_emb, wEmbT);

  // 2. cast projection weights to bf16
  cast_kernel<<<(kAttn * kEmb + 255) / 256, 256, 0, stream>>>(W_q, wqB, kAttn * kEmb);
  cast_kernel<<<(kAttn * kEmb + 255) / 256, 256, 0, stream>>>(W_k, wkB, kAttn * kEmb);
  cast_kernel<<<(kEmb * kEmb + 255) / 256, 256, 0, stream>>>(W_v, wvB, kEmb * kEmb);

  // 3. materialize bf16 embeddings (tokens + context) once
  embed_kernel<<<kRowsAll / 2, 256, 0, stream>>>(tokens, context, wEmbT, W_pos, embA);

  // 4. register-blocked WMMA projection GEMMs
  // q = emb @ W_q^T + b_q       -> [B*S, 64] row-major
  gemm_kernel<<<(kRowsQ / 16) * (kAttn / 64) / 8, 256, 0, stream>>>(
      embQ, wqB, b_q, qB, kRowsQ, kAttn, 0);
  // k = ctx_emb @ W_k^T + b_k   -> [C*B*S, 64] row-major
  gemm_kernel<<<(kRowsC / 16) * (kAttn / 64) / 8, 256, 0, stream>>>(
      embC, wkB, b_k, kB, kRowsC, kAttn, 0);
  // v = ctx_emb @ W_v^T + b_v   -> transposed [E, C*B*S]
  gemm_kernel<<<(kRowsC / 16) * (kEmb / 64) / 8, 256, 0, stream>>>(
      embC, wvB, b_v, vT, kRowsC, kEmb, 1);

  // 5. two-pass flash attention (WMMA scores + WMMA PV)
  attn_kernel<<<kCtx * kBatch * (kSeq / 16), 256, 0, stream>>>(qB, kB, vT, out);
}
